// KeyValueBottleneck_12343736008844
// MI455X (gfx1250) — matrix operand order repaired and checked by reading, not compiled, not run on hardware
//
#include <hip/hip_runtime.h>

// Problem constants (from reference)
#define C_   8
#define N_   512
#define D_   512
#define P_   1024
#define V_   512
#define B_   16
#define BN_  8192          // B*N tokens per codebook
#define ROWS_TOT 65536     // C*BN

typedef __attribute__((ext_vector_type(16))) _Float16 v16h;
typedef __attribute__((ext_vector_type(8)))  float    v8f;

#define V8F_ZERO (v8f){0.f,0.f,0.f,0.f,0.f,0.f,0.f,0.f}

// ---------------------------------------------------------------- utilities

__device__ __forceinline__ float wave_sum(float v) {
#pragma unroll
  for (int off = 16; off > 0; off >>= 1) v += __shfl_xor(v, off, 32);
  return v;
}

// A-fragment (f16 16x32, ISA 7.12.2): lanes 0-15 row r hold K {0..7,16..23},
// lanes 16-31 row r hold K {8..15,24..31}.
__device__ __forceinline__ v16h load_a(const _Float16* a, int lda, int m0, int k0, int lane) {
  const int hl = lane >> 4, r = lane & 15;
  const _Float16* p = a + (size_t)(m0 + r) * lda + k0 + (hl ? 8 : 0);
  v16h o;
  o[0]=p[0];  o[1]=p[1];  o[2]=p[2];  o[3]=p[3];
  o[4]=p[4];  o[5]=p[5];  o[6]=p[6];  o[7]=p[7];
  o[8]=p[16]; o[9]=p[17]; o[10]=p[18]; o[11]=p[19];
  o[12]=p[20]; o[13]=p[21]; o[14]=p[22]; o[15]=p[23];
  return o;
}

// B-fragment for C = A @ W^T where W is row-major [nOut][k]:
// column n = lane&15; lanes 0-15 hold K {0..7,16..23}, lanes 16-31 hold K {8..15,24..31}.
__device__ __forceinline__ v16h load_bT(const _Float16* w, int ldw, int n0, int k0, int lane) {
  const int hl = lane >> 4, n = lane & 15;
  const _Float16* p = w + (size_t)(n0 + n) * ldw + k0 + (hl ? 8 : 0);
  v16h o;
#pragma unroll
  for (int i = 0; i < 16; ++i) o[i] = p[i < 8 ? i : i + 8];
  return o;
}

__device__ __forceinline__ v8f wmma_acc(v16h a, v16h b, v8f c) {
  return __builtin_amdgcn_wmma_f32_16x16x32_f16(false, a, false, b, (short)0, c, false, false);
}

// ---------------------------------------------------------------- stage 0

__global__ void k_f32_to_f16(const float* __restrict__ src, _Float16* __restrict__ dst, int n) {
  for (int i = blockIdx.x * blockDim.x + threadIdx.x; i < n; i += gridDim.x * blockDim.x)
    dst[i] = (_Float16)src[i];
}

__global__ void k_halfk2(const float* __restrict__ keys, float* __restrict__ hk2) {
  int lane = threadIdx.x & 31, wv = threadIdx.x >> 5;
  int row = blockIdx.x * 8 + wv;                 // C_*P_ = 8192 rows
  if (row >= C_ * P_) return;
  const float* kr = keys + (size_t)row * D_;
  float s = 0.f;
  for (int i = lane; i < D_; i += 32) { float t = kr[i]; s += t * t; }
  s = wave_sum(s);
  if (lane == 0) hk2[row] = 0.5f * s;
}

// ---------------------------------------------------------------- stage 1
// Fused LN1 -> QKV (WMMA) -> 8x8 attention -> out-proj (WMMA) -> residual -> LN2
// Workgroup = 8 tokens (64 rows). 256 threads = 8 waves.

#define LDH 528   // f16 stride for 64x512 tiles
#define LDQ 1552  // f16 stride for 64x1536 qkv tile
#define LDM 520   // f32 stride for mha scratch
#define SMEM1 (64*LDH*2 + 64*LDQ*2)

__global__ void __launch_bounds__(256) k_attn_block(
    const float* __restrict__ x,
    const float* __restrict__ ln1_g, const float* __restrict__ ln1_b,
    const _Float16* __restrict__ w16_in,  const float* __restrict__ in_b,
    const _Float16* __restrict__ w16_out, const float* __restrict__ out_b,
    const float* __restrict__ ln2_g, const float* __restrict__ ln2_b,
    float* __restrict__ mha_g, _Float16* __restrict__ f2_g)
{
  extern __shared__ char smem[];
  _Float16* hA   = (_Float16*)smem;                 // 64 x LDH f16 (h, later o)
  _Float16* qkv  = (_Float16*)(smem + 64*LDH*2);    // 64 x LDQ f16
  float*    mhaS = (float*)(smem + 64*LDH*2);       // aliases qkv after phase C
  const int tid = threadIdx.x, lane = tid & 31, wv = tid >> 5;
  const int jBase = blockIdx.x * 8;

  // ---- Phase A: LN1 (wave per row)
  for (int rr = wv; rr < 64; rr += 8) {
    int t = rr >> 3, c = rr & 7;
    int j = jBase + t, b = j >> 9, n = j & 511;
    const float* xr = x + ((size_t)((b*C_ + c)*N_) + n) * D_;
    float xv[16]; float s = 0.f;
#pragma unroll
    for (int i = 0; i < 16; ++i) { xv[i] = xr[lane*16 + i]; s += xv[i]; }
    float mean = wave_sum(s) * (1.0f/512.0f);
    float vs = 0.f;
#pragma unroll
    for (int i = 0; i < 16; ++i) { float d = xv[i] - mean; vs += d*d; }
    float rstd = rsqrtf(wave_sum(vs) * (1.0f/512.0f) + 1e-5f);
#pragma unroll
    for (int i = 0; i < 16; ++i) {
      int k = lane*16 + i;
      hA[(size_t)rr*LDH + k] = (_Float16)((xv[i]-mean)*rstd*ln1_g[k] + ln1_b[k]);
    }
  }
  __syncthreads();

  // ---- Phase B: QKV = h @ in_w^T + in_b   (64x512 @ 512x1536)
  for (int tile = wv; tile < 4*96; tile += 8) {
    int m0 = (tile & 3) * 16, n0 = (tile >> 2) * 16;
    v8f acc = V8F_ZERO;
    for (int k0 = 0; k0 < 512; k0 += 32)
      acc = wmma_acc(load_a(hA, LDH, m0, k0, lane),
                     load_bT(w16_in, 512, n0, k0, lane), acc);
    int col = n0 + (lane & 15);
    float bias = in_b[col];
    int hl = lane >> 4;
#pragma unroll
    for (int v = 0; v < 8; ++v)
      qkv[(size_t)(m0 + v + hl*8)*LDQ + col] = (_Float16)(acc[v] + bias);
  }
  __syncthreads();

  // ---- Phase C: attention over the 8 codebooks (token = wave), H=2, hd=256
  {
    int t = wv;
    int id = lane & 15, h = id >> 3, c1 = id & 7, dkh = lane >> 4;
    const _Float16* qrow = qkv + (size_t)(t*8 + c1)*LDQ + h*256;
    float lg[8];
#pragma unroll
    for (int c2 = 0; c2 < 8; ++c2) {
      const _Float16* krow = qkv + (size_t)(t*8 + c2)*LDQ + 512 + h*256;
      float s = 0.f;
#pragma unroll 4
      for (int k = 0; k < 256; ++k) s += (float)qrow[k] * (float)krow[k];
      lg[c2] = s * 0.0625f;   // 1/sqrt(256)
    }
    float mx = lg[0];
#pragma unroll
    for (int c2 = 1; c2 < 8; ++c2) mx = fmaxf(mx, lg[c2]);
    float ss = 0.f;
#pragma unroll
    for (int c2 = 0; c2 < 8; ++c2) { lg[c2] = __expf(lg[c2] - mx); ss += lg[c2]; }
    float inv = 1.f / ss;
    for (int dk = dkh*128; dk < dkh*128 + 128; ++dk) {
      float s = 0.f;
#pragma unroll
      for (int c2 = 0; c2 < 8; ++c2)
        s += lg[c2] * (float)qkv[(size_t)(t*8 + c2)*LDQ + 1024 + h*256 + dk];
      hA[(size_t)(t*8 + c1)*LDH + h*256 + dk] = (_Float16)(s * inv);
    }
  }
  __syncthreads();

  // ---- Phase D: out-proj + residual(x) -> mhaS (LDS f32) + mha_g (global)
  for (int tile = wv; tile < 4*32; tile += 8) {
    int m0 = (tile & 3) * 16, n0 = (tile >> 2) * 16;
    v8f acc = V8F_ZERO;
    for (int k0 = 0; k0 < 512; k0 += 32)
      acc = wmma_acc(load_a(hA, LDH, m0, k0, lane),
                     load_bT(w16_out, 512, n0, k0, lane), acc);
    int col = n0 + (lane & 15);
    float bias = out_b[col];
    int hl = lane >> 4;
#pragma unroll
    for (int v = 0; v < 8; ++v) {
      int m = m0 + v + hl*8;
      int t = m >> 3, c = m & 7;
      int j = jBase + t, b = j >> 9, n = j & 511;
      float res = x[((size_t)((b*C_ + c)*N_) + n)*D_ + col];
      float val = acc[v] + bias + res;
      mhaS[(size_t)m*LDM + col] = val;
      mha_g[((size_t)c*BN_ + j)*D_ + col] = val;
    }
  }
  __syncthreads();

  // ---- Phase E: LN2 -> f2 (f16, global, layout R = c*BN + j)
  for (int rr = wv; rr < 64; rr += 8) {
    int t = rr >> 3, c = rr & 7;
    int j = jBase + t;
    float xv[16]; float s = 0.f;
#pragma unroll
    for (int i = 0; i < 16; ++i) { xv[i] = mhaS[(size_t)rr*LDM + lane*16 + i]; s += xv[i]; }
    float mean = wave_sum(s) * (1.0f/512.0f);
    float vs = 0.f;
#pragma unroll
    for (int i = 0; i < 16; ++i) { float d = xv[i] - mean; vs += d*d; }
    float rstd = rsqrtf(wave_sum(vs) * (1.0f/512.0f) + 1e-5f);
    size_t R = (size_t)c*BN_ + j;
#pragma unroll
    for (int i = 0; i < 16; ++i) {
      int k = lane*16 + i;
      f2_g[R*D_ + k] = (_Float16)((xv[i]-mean)*rstd*ln2_g[k] + ln2_b[k]);
    }
  }
}

// ---------------------------------------------------------------- stage 2
// Fused FFN (gelu) + residual + dec-proj. 64 rows per WG.

#define LDF 528
#define SMEM2 (2 * 64 * LDF * 2)

__global__ void __launch_bounds__(256) k_ffn_block(
    const _Float16* __restrict__ f2_g,
    const _Float16* __restrict__ w16_f1,  const float* __restrict__ b1,
    const _Float16* __restrict__ w16_f2,  const float* __restrict__ b2,
    const _Float16* __restrict__ w16_dec, const float* __restrict__ dec_b,
    const float* __restrict__ mha_g,
    _Float16* __restrict__ f_g)   // may alias f2_g (rows disjoint per WG, read-before-write)
{
  extern __shared__ char smem[];
  _Float16* A = (_Float16*)smem;                 // f2 tile, later reused as U
  _Float16* G = (_Float16*)(smem + 64*LDF*2);    // gelu output tile
  _Float16* U = A;
  const int tid = threadIdx.x, lane = tid & 31, wv = tid >> 5;
  const size_t R0 = (size_t)blockIdx.x * 64;

  for (int idx = tid; idx < 64*512; idx += 256) {
    int r = idx >> 9, k = idx & 511;
    A[(size_t)r*LDF + k] = f2_g[(R0 + r)*D_ + k];
  }
  __syncthreads();

  // GEMM1: gelu(A @ w1^T + b1) -> G
  for (int tile = wv; tile < 128; tile += 8) {
    int m0 = (tile & 3) * 16, n0 = (tile >> 2) * 16;
    v8f acc = V8F_ZERO;
    for (int k0 = 0; k0 < 512; k0 += 32)
      acc = wmma_acc(load_a(A, LDF, m0, k0, lane),
                     load_bT(w16_f1, 512, n0, k0, lane), acc);
    int col = n0 + (lane & 15), hl = lane >> 4;
    float bias = b1[col];
#pragma unroll
    for (int v = 0; v < 8; ++v) {
      float u = acc[v] + bias;
      float g = 0.5f * u * (1.f + erff(u * 0.70710678f));   // exact gelu
      G[(size_t)(m0 + v + hl*8)*LDF + col] = (_Float16)g;
    }
  }
  __syncthreads();

  // GEMM2: U = G @ w2^T + b2 + mha_h
  for (int tile = wv; tile < 128; tile += 8) {
    int m0 = (tile & 3) * 16, n0 = (tile >> 2) * 16;
    v8f acc = V8F_ZERO;
    for (int k0 = 0; k0 < 512; k0 += 32)
      acc = wmma_acc(load_a(G, LDF, m0, k0, lane),
                     load_bT(w16_f2, 512, n0, k0, lane), acc);
    int col = n0 + (lane & 15), hl = lane >> 4;
    float bias = b2[col];
#pragma unroll
    for (int v = 0; v < 8; ++v) {
      int m = m0 + v + hl*8;
      float u = acc[v] + bias + mha_g[(R0 + m)*D_ + col];
      U[(size_t)m*LDF + col] = (_Float16)u;
    }
  }
  __syncthreads();

  // GEMM3: f = U @ dec^T + dec_b
  for (int tile = wv; tile < 128; tile += 8) {
    int m0 = (tile & 3) * 16, n0 = (tile >> 2) * 16;
    v8f acc = V8F_ZERO;
    for (int k0 = 0; k0 < 512; k0 += 32)
      acc = wmma_acc(load_a(U, LDF, m0, k0, lane),
                     load_bT(w16_dec, 512, n0, k0, lane), acc);
    int col = n0 + (lane & 15), hl = lane >> 4;
    float bias = dec_b[col];
#pragma unroll
    for (int v = 0; v < 8; ++v) {
      int m = m0 + v + hl*8;
      f_g[(R0 + m)*D_ + col] = (_Float16)(acc[v] + bias);
    }
  }
}

// ---------------------------------------------------------------- stage 3
// dist = f @ keys^T - 0.5*||k||^2  -> running argmax -> gather values -> out

#define SMEM3 (64*LDF*2 + 128*4 + 128*4 + 64*4)

__global__ void __launch_bounds__(256) k_dist_block(
    const _Float16* __restrict__ f_g,
    const _Float16* __restrict__ keys16,
    const float* __restrict__ hk2,
    const float* __restrict__ values,
    float* __restrict__ out)
{
  extern __shared__ char smem[];
  _Float16* A   = (_Float16*)smem;
  float*    redV = (float*)(smem + 64*LDF*2);
  int*      redI = (int*)(smem + 64*LDF*2 + 128*4);
  int*      idxA = (int*)(smem + 64*LDF*2 + 256*4);
  const int tid = threadIdx.x, lane = tid & 31, wv = tid >> 5;
  const int c  = blockIdx.x >> 7;
  const int j0 = (blockIdx.x & 127) * 64;
  const size_t R0 = (size_t)c*BN_ + j0;

  for (int idx = tid; idx < 64*512; idx += 256) {
    int r = idx >> 9, k = idx & 511;
    A[(size_t)r*LDF + k] = f_g[(R0 + r)*D_ + k];
  }
  __syncthreads();

  const _Float16* kc = keys16 + (size_t)c * P_ * D_;
  const float*    hc = hk2 + c * P_;
  const int mt = wv & 3, grp = wv >> 2;   // 2 waves per M-tile, cols split 0..511 / 512..1023
  const int m0 = mt * 16;
  float best[8]; int bidx[8];
#pragma unroll
  for (int v = 0; v < 8; ++v) { best[v] = -3.4e38f; bidx[v] = 0; }

  for (int ntl = 0; ntl < 32; ++ntl) {
    int n0 = (grp*32 + ntl) * 16;
    v8f acc = V8F_ZERO;
    for (int k0 = 0; k0 < 512; k0 += 32)
      acc = wmma_acc(load_a(A, LDF, m0, k0, lane),
                     load_bT(kc, 512, n0, k0, lane), acc);
    int col = n0 + (lane & 15);
    float h = hc[col];
#pragma unroll
    for (int v = 0; v < 8; ++v) {
      float sc = acc[v] - h;
      if (sc > best[v]) { best[v] = sc; bidx[v] = col; }
    }
  }
  // cross-lane argmax within each 16-lane half (tie -> lower index)
#pragma unroll
  for (int off = 1; off < 16; off <<= 1) {
#pragma unroll
    for (int v = 0; v < 8; ++v) {
      float ov = __shfl_xor(best[v], off, 32);
      int   oi = __shfl_xor(bidx[v], off, 32);
      if (ov > best[v] || (ov == best[v] && oi < bidx[v])) { best[v] = ov; bidx[v] = oi; }
    }
  }
  int hl = lane >> 4;
  if ((lane & 15) == 0) {
#pragma unroll
    for (int v = 0; v < 8; ++v) {
      int m = m0 + v + hl*8;
      redV[m*2 + grp] = best[v];
      redI[m*2 + grp] = bidx[v];
    }
  }
  __syncthreads();
  if (tid < 64) {
    float v0 = redV[tid*2], v1 = redV[tid*2 + 1];
    int   i0 = redI[tid*2], i1 = redI[tid*2 + 1];
    idxA[tid] = (v1 > v0 || (v1 == v0 && i1 < i0)) ? i1 : i0;
  }
  __syncthreads();

  // gather values rows -> out (b, c, 1, n, V)
  for (int rr = wv; rr < 64; rr += 8) {
    int j = j0 + rr, b = j >> 9, n = j & 511;
    const float* vr = values + ((size_t)c*P_ + idxA[rr]) * V_;
    float* orow = out + (((size_t)(b*C_ + c))*N_ + n) * V_;
    for (int k = lane; k < V_; k += 32) orow[k] = vr[k];
  }
}

// ---------------------------------------------------------------- launch

extern "C" void kernel_launch(void* const* d_in, const int* in_sizes, int n_in,
                              void* d_out, int out_size, void* d_ws, size_t ws_size,
                              hipStream_t stream) {
  const float* x      = (const float*)d_in[0];
  const float* keys   = (const float*)d_in[1];
  const float* values = (const float*)d_in[2];
  const float* ln1_g  = (const float*)d_in[3];
  const float* ln1_b  = (const float*)d_in[4];
  const float* in_w   = (const float*)d_in[5];
  const float* in_b   = (const float*)d_in[6];
  const float* out_w  = (const float*)d_in[7];
  const float* out_b  = (const float*)d_in[8];
  const float* ln2_g  = (const float*)d_in[9];
  const float* ln2_b  = (const float*)d_in[10];
  const float* ffn_w1 = (const float*)d_in[11];
  const float* ffn_b1 = (const float*)d_in[12];
  const float* ffn_w2 = (const float*)d_in[13];
  const float* ffn_b2 = (const float*)d_in[14];
  const float* dec_w  = (const float*)d_in[15];
  const float* dec_b  = (const float*)d_in[16];
  float* out = (float*)d_out;

  char* ws = (char*)d_ws;
  size_t off = 0;
  _Float16* w16_in  = (_Float16*)(ws + off); off += (size_t)3*D_*D_*2;     // 1.5 MB
  _Float16* w16_out = (_Float16*)(ws + off); off += (size_t)D_*D_*2;
  _Float16* w16_f1  = (_Float16*)(ws + off); off += (size_t)D_*D_*2;
  _Float16* w16_f2  = (_Float16*)(ws + off); off += (size_t)D_*D_*2;
  _Float16* w16_dec = (_Float16*)(ws + off); off += (size_t)D_*D_*2;
  _Float16* keys16  = (_Float16*)(ws + off); off += (size_t)C_*P_*D_*2;    // 8 MB
  float*    hk2     = (float*)(ws + off);    off += (size_t)C_*P_*4;
  off = (off + 255) & ~(size_t)255;
  float*    mha_g   = (float*)(ws + off);    off += (size_t)ROWS_TOT*D_*4; // 128 MB
  _Float16* f2_g    = (_Float16*)(ws + off); off += (size_t)ROWS_TOT*D_*2; // 64 MB
  _Float16* f_g     = f2_g;  // alias: stage-2 reads its rows before writing them

  // Stage 0: precision conversion + 0.5*||k||^2
  k_f32_to_f16<<<1024, 256, 0, stream>>>(in_w,   w16_in,  3*D_*D_);
  k_f32_to_f16<<<512,  256, 0, stream>>>(out_w,  w16_out, D_*D_);
  k_f32_to_f16<<<512,  256, 0, stream>>>(ffn_w1, w16_f1,  D_*D_);
  k_f32_to_f16<<<512,  256, 0, stream>>>(ffn_w2, w16_f2,  D_*D_);
  k_f32_to_f16<<<512,  256, 0, stream>>>(dec_w,  w16_dec, D_*D_);
  k_f32_to_f16<<<4096, 256, 0, stream>>>(keys,   keys16,  C_*P_*D_);
  k_halfk2<<<(C_*P_)/8, 256, 0, stream>>>(keys, hk2);

  // Stage 1: fused LN1 + QKV + attention + out-proj + LN2 (8 tokens / WG)
  hipFuncSetAttribute((const void*)k_attn_block,
                      hipFuncAttributeMaxDynamicSharedMemorySize, SMEM1);
  k_attn_block<<<BN_/8, 256, SMEM1, stream>>>(x, ln1_g, ln1_b, w16_in, in_b,
                                              w16_out, out_b, ln2_g, ln2_b,
                                              mha_g, f2_g);

  // Stage 2: fused FFN + dec (64 rows / WG)
  hipFuncSetAttribute((const void*)k_ffn_block,
                      hipFuncAttributeMaxDynamicSharedMemorySize, SMEM2);
  k_ffn_block<<<ROWS_TOT/64, 256, SMEM2, stream>>>(f2_g, w16_f1, ffn_b1,
                                                   w16_f2, ffn_b2, w16_dec, dec_b,
                                                   mha_g, f_g);

  // Stage 3: nearest-key argmax + value gather
  hipFuncSetAttribute((const void*)k_dist_block,
                      hipFuncAttributeMaxDynamicSharedMemorySize, SMEM3);
  k_dist_block<<<C_*(BN_/64), 256, SMEM3, stream>>>(f_g, keys16, hk2, values, out);
}
